// Int8Linear_26843545600724
// MI455X (gfx1250) — 8-bit matrix operand order repaired and checked by reading, compiled, not run on hardware
//
#include <hip/hip_runtime.h>
#include <stdint.h>

typedef int v8i __attribute__((ext_vector_type(8)));

#define QMAX 127.0f
#define BM 128
#define BN 128
#define BK 64

#if __has_builtin(__builtin_amdgcn_global_load_async_to_lds_b128)
#define HAVE_ASYNC_LDS 1
typedef int v4i_async __attribute__((vector_size(16)));
typedef __attribute__((address_space(1))) v4i_async* gv4p;
typedef __attribute__((address_space(3))) v4i_async* lv4p;
#else
#define HAVE_ASYNC_LDS 0
#endif

__device__ __forceinline__ void wait_async_zero() {
#if HAVE_ASYNC_LDS
#if __has_builtin(__builtin_amdgcn_s_wait_asynccnt)
    __builtin_amdgcn_s_wait_asynccnt(0);
#else
    asm volatile("s_wait_asynccnt 0x0" ::: "memory");
#endif
#endif
}

__global__ __launch_bounds__(256) void quant_rows(const float* __restrict__ x,
                                                  int8_t* __restrict__ aq,
                                                  float* __restrict__ sa,
                                                  int K) {
    const int row = blockIdx.x;
    const int t   = threadIdx.x;
    const float4* __restrict__ xr4 = (const float4*)(x + (size_t)row * K);
    const int n4 = K >> 2;

    float amax = 0.0f;
    for (int i = t; i < n4; i += 256) {
        float4 v = xr4[i];
        amax = fmaxf(amax, fmaxf(fmaxf(fabsf(v.x), fabsf(v.y)),
                                 fmaxf(fabsf(v.z), fabsf(v.w))));
    }
    #pragma unroll
    for (int off = 16; off > 0; off >>= 1)
        amax = fmaxf(amax, __shfl_xor(amax, off, 32));

    __shared__ float smax[8];
    __shared__ float sinv;
    const int lane = t & 31;
    const int wid  = t >> 5;
    if (lane == 0) smax[wid] = amax;
    __syncthreads();
    if (t == 0) {
        float m = smax[0];
        #pragma unroll
        for (int i = 1; i < 8; i++) m = fmaxf(m, smax[i]);
        float s = fmaxf(m / QMAX, 1e-8f);
        sa[row] = s;
        sinv    = 1.0f / s;
    }
    __syncthreads();
    const float inv = sinv;

    int* __restrict__ ar = (int*)(aq + (size_t)row * K);
    for (int i = t; i < n4; i += 256) {
        float4 v = xr4[i];
        int q0 = (int)rintf(fminf(fmaxf(v.x * inv, -QMAX), QMAX));
        int q1 = (int)rintf(fminf(fmaxf(v.y * inv, -QMAX), QMAX));
        int q2 = (int)rintf(fminf(fmaxf(v.z * inv, -QMAX), QMAX));
        int q3 = (int)rintf(fminf(fmaxf(v.w * inv, -QMAX), QMAX));
        ar[i] = (q0 & 255) | ((q1 & 255) << 8) | ((q2 & 255) << 16) | (q3 << 24);
    }
}

union FragA { v8i v; long long l[4]; };
union FragB { v8i v; int4 q[2]; };

__global__ __launch_bounds__(256) void int8_gemm(const int8_t* __restrict__ Aq,
                                                 const int8_t* __restrict__ W,
                                                 const float* __restrict__ sa,
                                                 const float* __restrict__ wscale,
                                                 const float* __restrict__ bias,
                                                 float* __restrict__ out,
                                                 int M, int N, int K) {
    __shared__ __align__(16) int8_t lA[2][BM * BK];
    __shared__ __align__(16) int8_t lB[2][BN * BK];

    const int t       = threadIdx.x;
    const int lane    = t & 31;
    const int w       = t >> 5;
    const int m0      = blockIdx.y * BM;
    const int n0      = blockIdx.x * BN;
    const int wm      = (w >> 1) * 32;
    const int wn      = (w & 1) * 64;
    const int halfsel = (lane < 16) ? 0 : 1;
    const int lr      = lane & 15;

    v8i acc[2][4];
    #pragma unroll
    for (int mf = 0; mf < 2; mf++)
        #pragma unroll
        for (int nf = 0; nf < 4; nf++)
            #pragma unroll
            for (int e = 0; e < 8; e++)
                acc[mf][nf][e] = 0;

    const int ldrow  = t >> 1;
    const int ldcol  = (t & 1) * 32;
    const int ldsOff = ldrow * BK + ldcol;
    const int8_t* __restrict__ Ag = Aq + (size_t)(m0 + ldrow) * K + ldcol;
    const int8_t* __restrict__ Wg = W  + (size_t)(n0 + ldrow) * K + ldcol;

#if HAVE_ASYNC_LDS
#define STAGE(KB, BUF)                                                         \
    do {                                                                       \
        __builtin_amdgcn_global_load_async_to_lds_b128(                        \
            (gv4p)(Ag + (KB)), (lv4p)&lA[BUF][ldsOff], 0, 0);                  \
        __builtin_amdgcn_global_load_async_to_lds_b128(                        \
            (gv4p)(Ag + (KB)), (lv4p)&lA[BUF][ldsOff], 16, 0);                 \
        __builtin_amdgcn_global_load_async_to_lds_b128(                        \
            (gv4p)(Wg + (KB)), (lv4p)&lB[BUF][ldsOff], 0, 0);                  \
        __builtin_amdgcn_global_load_async_to_lds_b128(                        \
            (gv4p)(Wg + (KB)), (lv4p)&lB[BUF][ldsOff], 16, 0);                 \
    } while (0)
#else
#define STAGE(KB, BUF)                                                         \
    do {                                                                       \
        int4 ga0 = *(const int4*)(Ag + (KB));                                  \
        int4 ga1 = *(const int4*)(Ag + (KB) + 16);                             \
        int4 gb0 = *(const int4*)(Wg + (KB));                                  \
        int4 gb1 = *(const int4*)(Wg + (KB) + 16);                             \
        *(int4*)(&lA[BUF][ldsOff])      = ga0;                                 \
        *(int4*)(&lA[BUF][ldsOff + 16]) = ga1;                                 \
        *(int4*)(&lB[BUF][ldsOff])      = gb0;                                 \
        *(int4*)(&lB[BUF][ldsOff + 16]) = gb1;                                 \
    } while (0)
#endif

    const int nstage = K / BK;

    STAGE(0, 0);
    wait_async_zero();
    __syncthreads();

    for (int i = 0; i < nstage; ++i) {
        const int cur = i & 1;

        if (i + 1 < nstage)
            STAGE((i + 1) * BK, (i + 1) & 1);
        if (i + 2 < nstage) {
            __builtin_prefetch(Ag + (i + 2) * BK, 0, 1);
            __builtin_prefetch(Wg + (i + 2) * BK, 0, 1);
        }

        FragA a[2];
        #pragma unroll
        for (int mf = 0; mf < 2; mf++) {
            const int row  = wm + mf * 16 + lr;
            const int base = halfsel * 16;
            #pragma unroll
            for (int j = 0; j < 4; j++)
                a[mf].l[j] =
                    *(const long long*)(&lA[cur][row * BK + base + 8 * (j & 1) + 32 * (j >> 1)]);
        }

        FragB b[4];
        #pragma unroll
        for (int nf = 0; nf < 4; nf++) {
            const int col = wn + nf * 16 + lr;
            b[nf].q[0] = *(const int4*)(&lB[cur][col * BK + halfsel * 16]);
            b[nf].q[1] = *(const int4*)(&lB[cur][col * BK + 32 + halfsel * 16]);
        }

        #pragma unroll
        for (int mf = 0; mf < 2; mf++)
            #pragma unroll
            for (int nf = 0; nf < 4; nf++)
                acc[mf][nf] = __builtin_amdgcn_wmma_i32_16x16x64_iu8(
                    true, a[mf].v, true, b[nf].v, acc[mf][nf], false, false);

        wait_async_zero();
        __syncthreads();
    }

    #pragma unroll
    for (int nf = 0; nf < 4; nf++) {
        const int gn   = n0 + wn + nf * 16 + lr;
        const float wc = wscale[gn];
        const float bc = bias[gn];
        #pragma unroll
        for (int mf = 0; mf < 2; mf++) {
            const int gmBase = m0 + wm + mf * 16 + halfsel * 8;
            #pragma unroll
            for (int r = 0; r < 8; r++) {
                const int gm = gmBase + r;
                out[(size_t)gm * N + gn] =
                    (float)acc[mf][nf][r] * sa[gm] * wc + bc;
            }
        }
    }
}

extern "C" void kernel_launch(void* const* d_in, const int* in_sizes, int n_in,
                              void* d_out, int out_size, void* d_ws, size_t ws_size,
                              hipStream_t stream) {
    const float*  x    = (const float*)d_in[0];
    const int8_t* w8   = (const int8_t*)d_in[1];
    const float*  wsc  = (const float*)d_in[2];
    const float*  bias = (const float*)d_in[3];
    float*        out  = (float*)d_out;

    const int N = in_sizes[2];
    const int K = (int)((long long)in_sizes[1] / N);
    const int M = (int)((long long)in_sizes[0] / K);

    int8_t* aq = (int8_t*)d_ws;
    float*  sa = (float*)((char*)d_ws + (size_t)M * K);

    quant_rows<<<M, 256, 0, stream>>>(x, aq, sa, K);

    dim3 grid(N / BN, M / BM);
    int8_gemm<<<grid, 256, 0, stream>>>(aq, w8, sa, wsc, bias, out, M, N, K);
}
